// MultiHeadAttention_59974923321708
// MI455X (gfx1250) — matrix operand order repaired and checked by reading, not compiled, not run on hardware
//
#include <hip/hip_runtime.h>

// ---------------------------------------------------------------------------
// PVT-style spatial-reduction attention for MI455X (gfx1250, wave32, WMMA).
// All heavy math runs through v_wmma_f32_16x16x32_f16 with f32 accumulation.
// Shapes are fixed by the reference: B=2, N=4096, C=512, nh=8, hd=64, M=1024.
// ---------------------------------------------------------------------------

typedef __attribute__((ext_vector_type(16))) _Float16 v16h;
typedef __attribute__((ext_vector_type(8)))  _Float16 v8h;
typedef __attribute__((ext_vector_type(8)))  float    v8f;

#define WMMA(A, B, C) __builtin_amdgcn_wmma_f32_16x16x32_f16( \
    false, (A), false, (B), (short)0, (C), false, false)

// A-matrix 16x32 f16 fragment (per-lane): row = lane&15, half = lane>>4.
// element e: K = half*16 + e  -> one contiguous 16-elem chunk.
__device__ __forceinline__ v16h load_a_frag(const _Float16* row, int k0, int hf) {
  return *(const v16h*)(row + k0 + hf * 16);
}

// ---------------------------------------------------------------------------
// elementwise converts / rearranges
// ---------------------------------------------------------------------------
__global__ void cvt_f32_f16(const float* __restrict__ in, _Float16* __restrict__ out, int n) {
  int i = blockIdx.x * blockDim.x + threadIdx.x;
  if (i < n) out[i] = (_Float16)in[i];
}

// out[n*K + k] = in[k*N + n]   (store B transposed so B-frags load contiguously)
__global__ void transpose_cvt(const float* __restrict__ in, _Float16* __restrict__ out,
                              int K, int N) {
  int i = blockIdx.x * blockDim.x + threadIdx.x;
  if (i < K * N) {
    int k = i / N, n = i % N;
    out[(size_t)n * K + k] = (_Float16)in[i];
  }
}

// Wconv OIHW [512,512,2,2] -> WcT[co][kk], kk = (i*2+j)*512 + ci
__global__ void wconv_t(const float* __restrict__ in, _Float16* __restrict__ out, int n) {
  int i = blockIdx.x * blockDim.x + threadIdx.x;
  if (i < n) {
    int co = i >> 11;
    int r  = i & 2047;
    int ci = r >> 2;
    int ij = r & 3;
    out[((size_t)co << 11) + (ij << 9) + ci] = (_Float16)in[i];
  }
}

// Patch matrix for the stride-2 2x2 conv-as-GEMM:
// P[(b*1024 + m)*2048 + ij*512 + ci] = x[b, (2hp+i)*64 + (2wp+j), ci], m = hp*32+wp
__global__ void build_patches(const float* __restrict__ x, _Float16* __restrict__ P, int total) {
  int idx = blockIdx.x * blockDim.x + threadIdx.x;
  if (idx < total) {
    int ci = idx & 511;
    int ij = (idx >> 9) & 3;
    int m  = (idx >> 11) & 1023;
    int b  = idx >> 21;
    int i = ij >> 1, j = ij & 1;
    int hp = m >> 5, wp = m & 31;
    int r = (2 * hp + i) * 64 + (2 * wp + j);
    P[idx] = (_Float16)x[((size_t)b * 4096 + r) * 512 + ci];
  }
}

// ---------------------------------------------------------------------------
// Generic WMMA GEMM: C[row,col] = sum_k A[row,k] * BT[col,k] + bias[col]
// Each wave computes a 16x64 output strip (4 WMMA tiles), K-steps of 32.
// mode selects the epilogue layout permutation.
// ---------------------------------------------------------------------------
__global__ __launch_bounds__(256) void gemm16(
    const _Float16* __restrict__ A, const _Float16* __restrict__ BT,
    const float* __restrict__ bias, void* __restrict__ out,
    int Mrows, int N, int K, int mode) {
  int wave = blockIdx.x * (blockDim.x >> 5) + (threadIdx.x >> 5);
  int l  = threadIdx.x & 31;
  int lm = l & 15;
  int hf = l >> 4;
  int nQuads = N >> 6;
  int mTile = wave / nQuads;
  int nQuad = wave - mTile * nQuads;
  if (mTile * 16 >= Mrows) return;   // wave-uniform: EXEC stays all-ones
  int m0 = mTile * 16;
  int n0 = nQuad * 64;

  const _Float16* arow = A  + (size_t)(m0 + lm) * K;
  const _Float16* brow = BT + (size_t)(n0 + lm) * K;

  v8f acc0 = {}, acc1 = {}, acc2 = {}, acc3 = {};
  for (int k0 = 0; k0 < K; k0 += 32) {
    v16h a  = load_a_frag(arow, k0, hf);
    v16h b0 = *(const v16h*)(brow + k0 + hf * 16);
    v16h b1 = *(const v16h*)(brow + (size_t)16 * K + k0 + hf * 16);
    v16h b2 = *(const v16h*)(brow + (size_t)32 * K + k0 + hf * 16);
    v16h b3 = *(const v16h*)(brow + (size_t)48 * K + k0 + hf * 16);
    acc0 = WMMA(a, b0, acc0);
    acc1 = WMMA(a, b1, acc1);
    acc2 = WMMA(a, b2, acc2);
    acc3 = WMMA(a, b3, acc3);
  }

  v8f accs[4] = {acc0, acc1, acc2, acc3};
#pragma unroll
  for (int t = 0; t < 4; ++t) {
#pragma unroll
    for (int j = 0; j < 8; ++j) {
      int row = m0 + j + 8 * hf;
      int col = n0 + t * 16 + lm;
      float val = accs[t][j] + bias[col];
      if (mode == 0) {
        // plain row-major f32 (final projection -> d_out)
        ((float*)out)[(size_t)row * N + col] = val;
      } else if (mode == 1) {
        // Q: row = b*4096+n, col = h*64+d -> q[((b*8+h)*4096+n)*64+d] (f16)
        int b = row >> 12, n = row & 4095;
        int h = col >> 6,  d = col & 63;
        ((_Float16*)out)[(((size_t)b * 8 + h) * 4096 + n) * 64 + d] = (_Float16)val;
      } else if (mode == 2) {
        // conv: row = b*1024+m, col = co -> xc[(b*512+co)*1024+m] (f32, [B,C,M])
        int b = row >> 10, m = row & 1023;
        ((float*)out)[((size_t)b * 512 + col) * 1024 + m] = val;
      } else {
        // KV: row = b*512+c; col<512 -> K at [(b*8+h)*512+c]*64+d
        //                    col>=512 -> V stored transposed [(b*8+h)*64+d]*512+c
        int b = row >> 9, c = row & 511;
        _Float16* kbuf = (_Float16*)out;
        _Float16* vbuf = kbuf + (size_t)2 * 8 * 512 * 64;
        if (col < 512) {
          int h = col >> 6, d = col & 63;
          kbuf[(((size_t)b * 8 + h) * 512 + c) * 64 + d] = (_Float16)val;
        } else {
          int cc = col - 512;
          int h = cc >> 6, d = cc & 63;
          vbuf[(((size_t)b * 8 + h) * 64 + d) * 512 + c] = (_Float16)val;
        }
      }
    }
  }
}

// ---------------------------------------------------------------------------
// LayerNorm over last dim (M=1024) of xc [B*C, 1024]; gamma/beta per-position.
// ---------------------------------------------------------------------------
__global__ __launch_bounds__(256) void layernorm_k(
    const float* __restrict__ xc, const float* __restrict__ gamma,
    const float* __restrict__ beta, _Float16* __restrict__ out) {
  __shared__ float sh1[256];
  __shared__ float sh2[256];
  int row = blockIdx.x;
  const float* p = xc + (size_t)row * 1024;
  float s = 0.f, s2 = 0.f;
  for (int c = threadIdx.x; c < 1024; c += 256) {
    float v = p[c];
    s += v; s2 += v * v;
  }
  sh1[threadIdx.x] = s; sh2[threadIdx.x] = s2;
  __syncthreads();
  for (int off = 128; off > 0; off >>= 1) {
    if ((int)threadIdx.x < off) {
      sh1[threadIdx.x] += sh1[threadIdx.x + off];
      sh2[threadIdx.x] += sh2[threadIdx.x + off];
    }
    __syncthreads();
  }
  float mu  = sh1[0] * (1.f / 1024.f);
  float var = sh2[0] * (1.f / 1024.f) - mu * mu;
  float inv = rsqrtf(var + 1e-5f);
  for (int c = threadIdx.x; c < 1024; c += 256)
    out[(size_t)row * 1024 + c] = (_Float16)((p[c] - mu) * inv * gamma[c] + beta[c]);
}

// ---------------------------------------------------------------------------
// Attention: one wave per (b,h, 16 query rows). K-seq length = C = 512.
// Scores -> LDS(f32) -> softmax -> LDS(f16) -> P@V -> scrambled store.
// Output layout quirk: out2[b, h*512 + d*8 + (n>>9), n&511] = O[b,h,n,d]
// ---------------------------------------------------------------------------
__global__ __launch_bounds__(32) void attn_k(
    const _Float16* __restrict__ qb, const _Float16* __restrict__ kb,
    const _Float16* __restrict__ vT, _Float16* __restrict__ attn_out) {
  __shared__ float    S [16 * 512];
  __shared__ _Float16 Pb[16 * 512];

  int bh = blockIdx.x >> 8;     // b*8 + h
  int nt = blockIdx.x & 255;
  int n0 = nt * 16;
  int l  = threadIdx.x;
  int lm = l & 15;
  int hf = l >> 4;

  // Q fragments for the 16 query rows (K = hd = 64 -> two K-steps)
  const _Float16* qrow = qb + ((size_t)bh * 4096 + n0 + lm) * 64;
  v16h qa0 = load_a_frag(qrow, 0,  hf);
  v16h qa1 = load_a_frag(qrow, 32, hf);

  // scores: S[n, c] = (q . k) / sqrt(64)
  const _Float16* kbase = kb + (size_t)bh * 512 * 64;
  for (int ct = 0; ct < 32; ++ct) {
    int c0 = ct * 16;
    const _Float16* krow = kbase + (size_t)(c0 + lm) * 64;
    v16h b0 = *(const v16h*)(krow + hf * 16);
    v16h b1 = *(const v16h*)(krow + 32 + hf * 16);
    v8f acc = {};
    acc = WMMA(qa0, b0, acc);
    acc = WMMA(qa1, b1, acc);
#pragma unroll
    for (int j = 0; j < 8; ++j)
      S[(j + 8 * hf) * 512 + c0 + lm] = acc[j] * 0.125f;
  }
  __syncthreads();

  // softmax over 512 keys; lane pair (l, l^16) shares row lm
  {
    float* sr = S + lm * 512 + hf * 256;
    float mx = -3.4e38f;
    for (int c = 0; c < 256; ++c) mx = fmaxf(mx, sr[c]);
    mx = fmaxf(mx, __shfl_xor(mx, 16, 32));
    float sum = 0.f;
    for (int c = 0; c < 256; ++c) {
      float e = __expf(sr[c] - mx);
      sr[c] = e;
      sum += e;
    }
    sum += __shfl_xor(sum, 16, 32);
    float inv = 1.f / sum;
    _Float16* pr = Pb + lm * 512 + hf * 256;
    for (int c = 0; c < 256; ++c) pr[c] = (_Float16)(sr[c] * inv);
  }
  __syncthreads();

  // O = P @ V  (V pre-transposed: vT[bh, d, c], c contiguous)
  const _Float16* vbase = vT + (size_t)bh * 64 * 512;
  int b = bh >> 3, h = bh & 7;
#pragma unroll
  for (int dt = 0; dt < 4; ++dt) {
    int d0 = dt * 16;
    const _Float16* prow = Pb + lm * 512;
    const _Float16* vrow = vbase + (size_t)(d0 + lm) * 512;
    v8f acc = {};
    for (int k0 = 0; k0 < 512; k0 += 32) {
      v16h pa = load_a_frag(prow, k0, hf);
      v16h vb = *(const v16h*)(vrow + k0 + hf * 16);
      acc = WMMA(pa, vb, acc);
    }
#pragma unroll
    for (int j = 0; j < 8; ++j) {
      int n = n0 + j + 8 * hf;
      int d = d0 + lm;
      attn_out[((size_t)b * 4096 + h * 512 + d * 8 + (n >> 9)) * 512 + (n & 511)] =
          (_Float16)acc[j];
    }
  }
}

// ---------------------------------------------------------------------------
// host-side orchestration
// ---------------------------------------------------------------------------
extern "C" void kernel_launch(void* const* d_in, const int* in_sizes, int n_in,
                              void* d_out, int out_size, void* d_ws, size_t ws_size,
                              hipStream_t stream) {
  (void)in_sizes; (void)n_in; (void)out_size; (void)ws_size;

  const float* x      = (const float*)d_in[0];
  const float* Wq     = (const float*)d_in[1];
  const float* bq     = (const float*)d_in[2];
  const float* Wconv  = (const float*)d_in[3];
  const float* bconv  = (const float*)d_in[4];
  const float* gamma  = (const float*)d_in[5];
  const float* beta   = (const float*)d_in[6];
  const float* Wkv    = (const float*)d_in[7];
  const float* bkv    = (const float*)d_in[8];
  const float* Wproj  = (const float*)d_in[9];
  const float* bproj  = (const float*)d_in[10];

  // workspace carve-up (256B aligned)
  char* w = (char*)d_ws;
  size_t off = 0;
  auto carve = [&](size_t bytes) -> char* {
    char* p = w + off;
    off += (bytes + 255) & ~(size_t)255;
    return p;
  };
  _Float16* x_h    = (_Float16*)carve((size_t)8192 * 512 * 2);   // x in f16
  _Float16* WqT    = (_Float16*)carve((size_t)512 * 512 * 2);
  _Float16* WcT    = (_Float16*)carve((size_t)512 * 2048 * 2);
  _Float16* WkvT   = (_Float16*)carve((size_t)1024 * 1024 * 2);
  _Float16* WprojT = (_Float16*)carve((size_t)512 * 512 * 2);
  _Float16* q_h    = (_Float16*)carve((size_t)2 * 8 * 4096 * 64 * 2);
  _Float16* P      = (_Float16*)carve((size_t)2048 * 2048 * 2);
  float*    xc     = (float*)   carve((size_t)2 * 512 * 1024 * 4);
  _Float16* xln    = (_Float16*)carve((size_t)1024 * 1024 * 2);
  _Float16* k_h    = (_Float16*)carve((size_t)2 * 8 * 512 * 64 * 2);  // K then vT contiguous
  _Float16* vT_h   = (_Float16*)carve((size_t)2 * 8 * 64 * 512 * 2);
  _Float16* attn_h = (_Float16*)carve((size_t)8192 * 512 * 2);
  (void)vT_h;  // addressed as k_h + 2*8*512*64 inside gemm mode 3

  // 1) precision / layout conversions
  cvt_f32_f16  <<<16384, 256, 0, stream>>>(x, x_h, 8192 * 512);
  transpose_cvt<<<1024,  256, 0, stream>>>(Wq,    WqT,    512,  512);
  wconv_t      <<<4096,  256, 0, stream>>>(Wconv, WcT,    512 * 2048);
  transpose_cvt<<<4096,  256, 0, stream>>>(Wkv,   WkvT,   1024, 1024);
  transpose_cvt<<<1024,  256, 0, stream>>>(Wproj, WprojT, 512,  512);
  build_patches<<<16384, 256, 0, stream>>>(x, P, 2048 * 2048);

  // 2) Q = x @ Wq + bq  -> per-head layout (f16)
  gemm16<<<512, 256, 0, stream>>>(x_h, WqT, bq, q_h, 8192, 512, 512, 1);

  // 3) conv-as-GEMM -> xc [B, C, M] (f32)
  gemm16<<<128, 256, 0, stream>>>(P, WcT, bconv, xc, 2048, 512, 2048, 2);

  // 4) LayerNorm over M
  layernorm_k<<<1024, 256, 0, stream>>>(xc, gamma, beta, xln);

  // 5) KV = xln @ Wkv + bkv -> K [bh,c,d] and V transposed [bh,d,c]
  gemm16<<<128, 256, 0, stream>>>(xln, WkvT, bkv, k_h, 1024, 1024, 1024, 3);

  // 6) attention with softmax, scrambled output layout
  attn_k<<<4096, 32, 0, stream>>>(q_h, k_h, k_h + (size_t)2 * 8 * 512 * 64, attn_h);

  // 7) final projection -> d_out (f32)
  gemm16<<<512, 256, 0, stream>>>(attn_h, WprojT, bproj, (float*)d_out, 8192, 512, 512, 0);
}
